// GridToStation_68324339745130
// MI455X (gfx1250) — matrix operand order repaired and checked by reading, not compiled, not run on hardware
//
#include <hip/hip_runtime.h>
#include <hip/hip_bf16.h>
#include <math.h>

typedef __attribute__((ext_vector_type(16))) _Float16 v16h;
typedef __attribute__((ext_vector_type(8)))  _Float16 v8h;
typedef __attribute__((ext_vector_type(8)))  float    v8f;

#define C_DIM 256
#define H_DIM 721
#define W_DIM 1440

// ---------------------------------------------------------------------------
// fp32 -> fp16 weight conversion (W1/W2 are 256x256, trivial)
// ---------------------------------------------------------------------------
__global__ void f32_to_f16_kernel(const float* __restrict__ src,
                                  _Float16* __restrict__ dst, int n) {
    int i = blockIdx.x * blockDim.x + threadIdx.x;
    if (i < n) dst[i] = (_Float16)src[i];
}

// ---------------------------------------------------------------------------
// Bilinear sampling: one block (256 threads) per station, thread = channel.
// Loads are inherently scattered (channel stride = H*W*4B); the f16 store of
// x[n, 0..255] is fully coalesced (512B per block).
// ---------------------------------------------------------------------------
__global__ __launch_bounds__(256)
void bilinear_sample_kernel(const float* __restrict__ grid,
                            const float* __restrict__ coords,
                            _Float16* __restrict__ xout) {
    const int n = blockIdx.x;
    const int c = threadIdx.x;

    const float lat = coords[n * 3 + 0];
    const float lon = coords[n * 3 + 1];

    float fx = (lon * (1.0f / 180.0f) + 1.0f) * 0.5f * (float)(W_DIM - 1);
    float fy = (lat * (1.0f / 90.0f)  + 1.0f) * 0.5f * (float)(H_DIM - 1);
    fx = fminf(fmaxf(fx, 0.0f), (float)(W_DIM - 1));
    fy = fminf(fmaxf(fy, 0.0f), (float)(H_DIM - 1));

    const float fx0 = floorf(fx);
    const float fy0 = floorf(fy);
    const float wx = fx - fx0;
    const float wy = fy - fy0;
    const int ix0 = (int)fx0;
    const int iy0 = (int)fy0;
    const int ix1 = min(ix0 + 1, W_DIM - 1);
    const int iy1 = min(iy0 + 1, H_DIM - 1);

    const long HW = (long)H_DIM * (long)W_DIM;
    const float* __restrict__ g = grid + (long)c * HW;

    const float v00 = g[(long)iy0 * W_DIM + ix0];
    const float v01 = g[(long)iy0 * W_DIM + ix1];
    const float v10 = g[(long)iy1 * W_DIM + ix0];
    const float v11 = g[(long)iy1 * W_DIM + ix1];

    const float val = v00 * (1.0f - wx) * (1.0f - wy)
                    + v01 * wx          * (1.0f - wy)
                    + v10 * (1.0f - wx) * wy
                    + v11 * wx          * wy;

    xout[(long)n * C_DIM + c] = (_Float16)val;
}

// ---------------------------------------------------------------------------
// WMMA GEMM: D[M,N] = act( A[M,K] @ W[N,K]^T + bias[N] )
// Each wave (32 lanes) computes a 16(M) x 64(N) tile: 4 f32 accumulators,
// A-fragment reused across 4 B-fragments, K=256 fully unrolled (8 steps).
// Fragment layouts: 16-bit A 16x32 / B 32x16 in contiguous k = 16*hi + s order; per ISA 7.12.2 (
// 32-bit C/D 16x16).
// ---------------------------------------------------------------------------
template <bool GELU_ACT, typename OutT>
__global__ __launch_bounds__(256)
void gemm_wmma_kernel(const _Float16* __restrict__ A,   // M x K row-major (f16)
                      const _Float16* __restrict__ Bw,  // N x K row-major (f16), row j = weights of out col j
                      const float*    __restrict__ bias, // N (f32)
                      OutT* __restrict__ D,              // M x N row-major
                      int M, int N, int K) {
    const int wid  = blockIdx.x * 8 + (threadIdx.x >> 5);
    const int lane = threadIdx.x & 31;
    const int lo   = lane & 15;     // row/col within 16-wide tile
    const int hi   = lane >> 4;     // half-wave selector

    const int nstrips = N >> 6;                 // 64-col strips
    const int m0 = (wid / nstrips) << 4;
    const int n0 = (wid % nstrips) << 6;

    v8f acc[4] = {};

    // Per-lane base pointers (all 16B/32B aligned: K=256 halves -> 512B rows)
    const _Float16* __restrict__ arow = A  + (size_t)(m0 + lo) * K + 16 * hi;
    const _Float16* __restrict__ brow = Bw + (size_t)(n0 + lo) * K + 16 * hi;

#pragma unroll
    for (int kk = 0; kk < 256; kk += 32) {
        // A frag: elems 0..7 = K(kk+16hi .. +7), elems 8..15 = K(kk+16hi+8 .. +7)
        const v8h a0 = *(const v8h*)(arow + kk);
        const v8h a1 = *(const v8h*)(arow + kk + 8);
        const v16h afrag = __builtin_shufflevector(
            a0, a1, 0, 1, 2, 3, 4, 5, 6, 7, 8, 9, 10, 11, 12, 13, 14, 15);

#pragma unroll
        for (int t = 0; t < 4; ++t) {
            // B frag: 16 contiguous halves of W row (n0 + t*16 + lo) at K=kk+16hi
            const v16h bfrag = *(const v16h*)(brow + (size_t)t * 16 * K + kk);
            acc[t] = __builtin_amdgcn_wmma_f32_16x16x32_f16(
                /*neg_a=*/false, afrag, /*neg_b=*/false, bfrag,
                /*c_mod=*/(short)0, acc[t], /*reuse_a=*/false, /*reuse_b=*/false);
        }
    }

    // Epilogue: bias (+ exact GELU), fp32 math, store.
#pragma unroll
    for (int t = 0; t < 4; ++t) {
        const int ncol = n0 + t * 16 + lo;
        const float bv = bias[ncol];
#pragma unroll
        for (int r = 0; r < 8; ++r) {
            const int row = m0 + r + 8 * hi;
            float v = acc[t][r] + bv;
            if (GELU_ACT)
                v = 0.5f * v * (1.0f + erff(v * 0.70710678118654752f));
            D[(size_t)row * N + ncol] = (OutT)v;
        }
    }
}

// ---------------------------------------------------------------------------
// Launcher
// inputs: grid_features(f32), station_coords(f32), W1(f32), b1(f32), W2(f32), b2(f32)
// output: (N, C) f32
// ---------------------------------------------------------------------------
extern "C" void kernel_launch(void* const* d_in, const int* in_sizes, int n_in,
                              void* d_out, int out_size, void* d_ws, size_t ws_size,
                              hipStream_t stream) {
    const float* grid   = (const float*)d_in[0];
    const float* coords = (const float*)d_in[1];
    const float* W1     = (const float*)d_in[2];
    const float* b1     = (const float*)d_in[3];
    const float* W2     = (const float*)d_in[4];
    const float* b2     = (const float*)d_in[5];
    float* out = (float*)d_out;

    const int Nst = in_sizes[1] / 3;   // 16384 stations

    char* ws = (char*)d_ws;
    _Float16* x16 = (_Float16*)ws;                                  // N*C f16
    _Float16* h16 = (_Float16*)(ws + (size_t)Nst * C_DIM * 2);      // N*C f16
    _Float16* w1h = (_Float16*)(ws + (size_t)Nst * C_DIM * 4);      // C*C f16
    _Float16* w2h = w1h + (size_t)C_DIM * C_DIM;                    // C*C f16

    const int wn = C_DIM * C_DIM;
    f32_to_f16_kernel<<<(wn + 255) / 256, 256, 0, stream>>>(W1, w1h, wn);
    f32_to_f16_kernel<<<(wn + 255) / 256, 256, 0, stream>>>(W2, w2h, wn);

    bilinear_sample_kernel<<<Nst, 256, 0, stream>>>(grid, coords, x16);

    // (Nst/16) m-tiles * (C/64) n-strips waves, 8 waves per block
    const int gemm_blocks = (Nst / 16) * (C_DIM / 64) / 8;   // 512
    gemm_wmma_kernel<true, _Float16><<<gemm_blocks, 256, 0, stream>>>(
        x16, w1h, b1, h16, Nst, C_DIM, C_DIM);
    gemm_wmma_kernel<false, float><<<gemm_blocks, 256, 0, stream>>>(
        h16, w2h, b2, out, Nst, C_DIM, C_DIM);
}
